// AnalysisTransform_37666863186106
// MI455X (gfx1250) — compile-verified, matrix operand order checked by reading, not run on hardware
//
#include <hip/hip_runtime.h>

// ---------------------------------------------------------------------------
// Sparse voxel CNN encoder for MI455X (gfx1250, wave32, WMMA f16->f32).
// Occupancy-masked convs are truly sparse -> compact features per level,
// implicit-GEMM gather conv with v_wmma_f32_16x16x32_f16.
// One wave: 16 output voxels x COUT channels; per 32-K step: one A-fragment
// load (2x b128) reused for COUT/16 WMMAs against packed B fragments.
// Compacted features (<=19MB/level) are L2-resident (192MB L2).
// ---------------------------------------------------------------------------

typedef __attribute__((ext_vector_type(16))) _Float16 v16h;
typedef __attribute__((ext_vector_type(8)))  _Float16 v8h;
typedef __attribute__((ext_vector_type(8)))  float    v8f;

static constexpr int D0 = 128, D1 = 64, D2 = 32, D3 = 16;
static constexpr int NMAX0 = 150016;   // >= N_PTS, multiple of 16/256
static constexpr int NMAX2 = 32768;    // 32^3

// ---- workspace layout (byte offsets, all 256B aligned) --------------------
static constexpr size_t MAP0_OFF = 0;                                   // int[128^3]
static constexpr size_t MAP1_OFF = MAP0_OFF + (size_t)D0*D0*D0*4;       // int[64^3]
static constexpr size_t MAP2_OFF = MAP1_OFF + (size_t)D1*D1*D1*4;       // int[32^3]
static constexpr size_t CNT_OFF  = MAP2_OFF + (size_t)D2*D2*D2*4;       // int[64]
static constexpr size_t IDX0_OFF = CNT_OFF  + 256;                      // int[NMAX0]
static constexpr size_t IDX1_OFF = IDX0_OFF + (size_t)NMAX0*4;          // int[NMAX0]
static constexpr size_t IDX2_OFF = IDX1_OFF + (size_t)NMAX0*4;          // int[NMAX2]
static constexpr size_t ACC0_OFF = IDX2_OFF + (size_t)NMAX2*4;          // float[NMAX0*3]
static constexpr size_t F0IN_OFF = ACC0_OFF + (size_t)NMAX0*3*4;        // half[NMAX0*3]
static constexpr size_t F0A_OFF  = F0IN_OFF + (size_t)NMAX0*3*2;        // half[NMAX0*64]
static constexpr size_t F1A_OFF  = F0A_OFF  + (size_t)NMAX0*64*2;       // half[NMAX0*64]
static constexpr size_t F1B_OFF  = F1A_OFF  + (size_t)NMAX0*64*2;       // half[NMAX0*64]
static constexpr size_t F2A_OFF  = F1B_OFF  + (size_t)NMAX0*64*2;       // half[NMAX2*32]
static constexpr size_t F2B_OFF  = F2A_OFF  + (size_t)NMAX2*32*2;       // half[NMAX2*32]
static constexpr size_t WPK_OFF  = F2B_OFF  + (size_t)NMAX2*32*2;       // fragment-packed f16 weights
// fragment-packed weight sub-offsets in halves, K padded to 32 per layer
static constexpr size_t WP1 = 0;                    // 96  *64
static constexpr size_t WP2 = WP1 + 96   * 64;      // 1728*64
static constexpr size_t WP3 = WP2 + 1728 * 64;
static constexpr size_t WP4 = WP3 + 1728 * 64;      // 1728*32
static constexpr size_t WP5 = WP4 + 1728 * 32;      // 864 *32
static constexpr size_t WP6 = WP5 + 864  * 32;
static constexpr size_t WS_NEEDED = WPK_OFF + (WP6 + 864 * 32) * 2;     // ~73 MB

// ---------------------------------------------------------------------------
// Occupancy build kernels
// ---------------------------------------------------------------------------
__global__ void mark_points(const int* __restrict__ coords, int npts,
                            int* __restrict__ map0) {
  int p = blockIdx.x * blockDim.x + threadIdx.x;
  if (p >= npts) return;
  int i = coords[p * 3 + 0], j = coords[p * 3 + 1], k = coords[p * 3 + 2];
  map0[(i * D0 + j) * D0 + k] = 1;
}

__global__ void compact_map(int* __restrict__ map, int nvox,
                            int* __restrict__ counter, int* __restrict__ idx) {
  int v = blockIdx.x * blockDim.x + threadIdx.x;
  if (v >= nvox) return;
  if (map[v] != -1) {
    int row = atomicAdd(counter, 1);
    map[v] = row;
    idx[row] = v;
  }
}

__global__ void scatter_feats(const int* __restrict__ coords,
                              const float* __restrict__ feats, int npts,
                              const int* __restrict__ map0,
                              float* __restrict__ acc) {
  int p = blockIdx.x * blockDim.x + threadIdx.x;
  if (p >= npts) return;
  int i = coords[p * 3 + 0], j = coords[p * 3 + 1], k = coords[p * 3 + 2];
  int row = map0[(i * D0 + j) * D0 + k];
  atomicAdd(&acc[row * 3 + 0], feats[p * 3 + 0]);
  atomicAdd(&acc[row * 3 + 1], feats[p * 3 + 1]);
  atomicAdd(&acc[row * 3 + 2], feats[p * 3 + 2]);
}

template <int FD>
__global__ void mark_parents(const int* __restrict__ idx_fine,
                             const int* __restrict__ cnt, int slot,
                             int* __restrict__ map_coarse) {
  int r = blockIdx.x * blockDim.x + threadIdx.x;
  if (r >= cnt[slot]) return;
  int lin = idx_fine[r];
  int z = lin / (FD * FD);
  int rm = lin - z * FD * FD;
  int y = rm / FD;
  int x = rm - y * FD;
  constexpr int CD = FD / 2;
  map_coarse[((z >> 1) * CD + (y >> 1)) * CD + (x >> 1)] = 1;
}

__global__ void cvt_counted(const float* __restrict__ src,
                            _Float16* __restrict__ dst,
                            const int* __restrict__ cnt, int slot, int mult) {
  int i = blockIdx.x * blockDim.x + threadIdx.x;
  if (i >= cnt[slot] * mult) return;
  dst[i] = (_Float16)src[i];
}

// Repack weights [KT][COUT] f32 -> fragment-major f16 so each lane's 16
// B-halves for one 32-K block are a single contiguous 32B load.
// dst[((kb32*COUT + n)*32) + hi*16 + jj]  with jj=(v*2+e),
// kloc = (v&3)*2 + (v>>2)*16 + hi*8 + e  (ISA 16-bit 32x16 fragment layout).
__global__ void pack_w_frag(const float* __restrict__ src,
                            _Float16* __restrict__ dst, int KT, int COUT_) {
  int i = blockIdx.x * blockDim.x + threadIdx.x;
  int KP = (KT + 31) & ~31;
  if (i >= KP * COUT_) return;
  int kb32 = i / (COUT_ * 32);
  int rem = i - kb32 * COUT_ * 32;
  int n = rem >> 5;
  int j = rem & 31;
  int hi = j >> 4, jj = j & 15;
  int v = jj >> 1, e = jj & 1;
  int kloc = ((v & 3) * 2) + ((v >> 2) * 16) + hi * 8 + e;
  int k = kb32 * 32 + kloc;
  dst[i] = (k < KT) ? (_Float16)src[k * COUT_ + n] : (_Float16)0.f;
}

// ---------------------------------------------------------------------------
// Implicit-GEMM sparse conv. One wave computes a 16 x COUT output tile
// (16 compacted output voxels, all channel tiles), K in 32-steps via
// v_wmma_f32_16x16x32_f16. Neighbor compact-rows staged in LDS once; each
// A fragment is loaded once and reused for COUT/16 WMMAs.
// ---------------------------------------------------------------------------
template <int CIN, int COUT, int STRIDE, int IND, bool RELU, bool DENSE_OUT,
          bool F32OUT>
__global__ __launch_bounds__(32) void conv_wmma(
    const int* __restrict__ map_in, const _Float16* __restrict__ feat_in,
    const int* __restrict__ idx_out, const int* __restrict__ cnt, int cnt_slot,
    const _Float16* __restrict__ wpk, const float* __restrict__ bias,
    _Float16* __restrict__ feat_out, float* __restrict__ f32_out,
    const int* __restrict__ occ_map) {
  constexpr int OUTD = (STRIDE == 2) ? (IND / 2) : IND;
  constexpr int KT = 27 * CIN;
  constexpr int NKB = (KT + 31) / 32;
  constexpr int NT = COUT / 16;   // channel tiles per wave

  __shared__ int nbr[27 * 16];    // [tap][m] -> input compact row (-1 empty)

  const int nout = DENSE_OUT ? (OUTD * OUTD * OUTD) : cnt[cnt_slot];
  const int tile = blockIdx.x * 16;
  if (tile >= nout) return;  // uniform: EXEC stays all-ones for WMMA

  const int lane = threadIdx.x;
  const int mloc = lane & 15;
  const int hi = lane >> 4;

  // ---- stage 27 neighbor rows per output voxel into LDS -------------------
  for (int t = lane; t < 27 * 16; t += 32) {
    const int m = t & 15;
    const int tap = t >> 4;
    int r = -1;
    const int orow = tile + m;
    if (orow < nout) {
      const int ol = DENSE_OUT ? orow : idx_out[orow];
      const int oz = ol / (OUTD * OUTD);
      const int orem = ol - oz * OUTD * OUTD;
      const int oy = orem / OUTD;
      const int ox = orem - oy * OUTD;
      const int dz = tap / 9;
      const int t9 = tap - dz * 9;
      const int dy = t9 / 3;
      const int dx = t9 - dy * 3;
      const int iz = oz * STRIDE - 1 + dz;
      const int iy = oy * STRIDE - 1 + dy;
      const int ix = ox * STRIDE - 1 + dx;
      if ((unsigned)iz < (unsigned)IND && (unsigned)iy < (unsigned)IND &&
          (unsigned)ix < (unsigned)IND)
        r = map_in[(iz * IND + iy) * IND + ix];
    }
    nbr[t] = r;
  }
  __syncthreads();   // single-wave WG: lowers to s_nop; DS stays in-order

  v8f c[NT];
#pragma unroll
  for (int nt = 0; nt < NT; ++nt) c[nt] = (v8f){};

  if (CIN % 32 == 0) {
    // ---- fast path: one 32-K block == 32 contiguous channels of one tap ---
    constexpr int SUBS = (CIN >= 32) ? (CIN / 32) : 1;
    for (int tap = 0; tap < 27; ++tap) {
      const int r = nbr[tap * 16 + mloc];
#pragma unroll
      for (int sub = 0; sub < SUBS; ++sub) {
        const int kb32 = tap * SUBS + sub;
        v8h lo = {}, h8 = {};
        if (r >= 0) {
          const _Float16* base = feat_in + (size_t)r * CIN + sub * 32 + hi * 8;
          lo = *(const v8h*)(base);        // ci: sub*32 + hi*8 + 0..7
          h8 = *(const v8h*)(base + 16);   // ci: sub*32 + 16 + hi*8 + 0..7
        }
        v16h a;
#pragma unroll
        for (int j = 0; j < 8; ++j) {
          a[j] = lo[j];
          a[j + 8] = h8[j];
        }
        const _Float16* wb =
            wpk + ((size_t)kb32 * COUT + mloc) * 32 + hi * 16;
#pragma unroll
        for (int nt = 0; nt < NT; ++nt) {
          const v16h b = *(const v16h*)(wb + (size_t)nt * 16 * 32);
          c[nt] = __builtin_amdgcn_wmma_f32_16x16x32_f16(
              false, a, false, b, (short)0, c[nt], false, false);
        }
      }
    }
  } else {
    // ---- generic path (conv1, CIN=3, K=81 padded to 96) -------------------
    for (int kb = 0; kb < NKB; ++kb) {
      v16h a;
#pragma unroll
      for (int v = 0; v < 8; ++v) {
        const int kbase = kb * 32 + ((v & 3) * 2) + ((v >> 2) * 16) + hi * 8;
#pragma unroll
        for (int e = 0; e < 2; ++e) {
          const int k = kbase + e;
          _Float16 av = (_Float16)0.f;
          if (k < KT) {
            const int tap = k / CIN;
            const int ci = k - tap * CIN;
            const int r = nbr[tap * 16 + mloc];
            if (r >= 0) av = feat_in[r * CIN + ci];
          }
          a[v * 2 + e] = av;
        }
      }
      const _Float16* wb = wpk + ((size_t)kb * COUT + mloc) * 32 + hi * 16;
#pragma unroll
      for (int nt = 0; nt < NT; ++nt) {
        const v16h b = *(const v16h*)(wb + (size_t)nt * 16 * 32);
        c[nt] = __builtin_amdgcn_wmma_f32_16x16x32_f16(
            false, a, false, b, (short)0, c[nt], false, false);
      }
    }
  }

  // ---- epilogue: bias (+relu), f16 compact store or masked dense f32 ------
  float bn[NT];
#pragma unroll
  for (int nt = 0; nt < NT; ++nt) bn[nt] = bias[nt * 16 + mloc];

#pragma unroll
  for (int r = 0; r < 8; ++r) {
    const int M = r + hi * 8;            // C/D layout: VGPR r, N = lane&15
    const int orow = tile + M;
    if (orow < nout) {
      const int ol = (F32OUT || !DENSE_OUT)
                         ? (DENSE_OUT ? orow : (F32OUT ? idx_out[orow] : 0))
                         : 0;
      bool occ = true;
      if (DENSE_OUT) {   // occ3 mask: any occupied child at fine level
        const int z = ol / (OUTD * OUTD);
        const int rm = ol - z * OUTD * OUTD;
        const int y = rm / OUTD;
        const int x = rm - y * OUTD;
        occ = false;
#pragma unroll
        for (int cz = 0; cz < 2; ++cz)
#pragma unroll
          for (int cy = 0; cy < 2; ++cy)
#pragma unroll
            for (int cx = 0; cx < 2; ++cx)
              if (occ_map[((z * 2 + cz) * IND + (y * 2 + cy)) * IND +
                          (x * 2 + cx)] >= 0)
                occ = true;
      }
#pragma unroll
      for (int nt = 0; nt < NT; ++nt) {
        const int n = nt * 16 + mloc;
        float vv = c[nt][r] + bn[nt];
        if (RELU) vv = fmaxf(vv, 0.f);
        if (F32OUT) {
          f32_out[ol * COUT + n] = occ ? vv : 0.f;
        } else {
          feat_out[orow * COUT + n] = (_Float16)vv;
        }
      }
    }
  }
}

// ---------------------------------------------------------------------------
extern "C" void kernel_launch(void* const* d_in, const int* in_sizes, int n_in,
                              void* d_out, int out_size, void* d_ws,
                              size_t ws_size, hipStream_t stream) {
  (void)n_in; (void)out_size; (void)ws_size;
  const int* coords = (const int*)d_in[0];
  const float* feats = (const float*)d_in[1];
  const float* w[6];
  const float* b[6];
  for (int i = 0; i < 6; ++i) {
    w[i] = (const float*)d_in[2 + 2 * i];
    b[i] = (const float*)d_in[3 + 2 * i];
  }
  const int npts = in_sizes[0] / 3;

  char* ws = (char*)d_ws;
  int* map0 = (int*)(ws + MAP0_OFF);
  int* map1 = (int*)(ws + MAP1_OFF);
  int* map2 = (int*)(ws + MAP2_OFF);
  int* cnt  = (int*)(ws + CNT_OFF);     // [0]=N0 [1]=N1 [2]=N2
  int* idx0 = (int*)(ws + IDX0_OFF);
  int* idx1 = (int*)(ws + IDX1_OFF);
  int* idx2 = (int*)(ws + IDX2_OFF);
  float* acc0 = (float*)(ws + ACC0_OFF);
  _Float16* f0in = (_Float16*)(ws + F0IN_OFF);
  _Float16* f0a  = (_Float16*)(ws + F0A_OFF);
  _Float16* f1a  = (_Float16*)(ws + F1A_OFF);
  _Float16* f1b  = (_Float16*)(ws + F1B_OFF);
  _Float16* f2a  = (_Float16*)(ws + F2A_OFF);
  _Float16* f2b  = (_Float16*)(ws + F2B_OFF);
  _Float16* wpk  = (_Float16*)(ws + WPK_OFF);

  // init: maps = -1 (0xFF), counters/accumulator = 0   (graph-capturable)
  hipMemsetAsync(map0, 0xFF, (size_t)D0 * D0 * D0 * 4, stream);
  hipMemsetAsync(map1, 0xFF, (size_t)D1 * D1 * D1 * 4, stream);
  hipMemsetAsync(map2, 0xFF, (size_t)D2 * D2 * D2 * 4, stream);
  hipMemsetAsync(cnt, 0, 256, stream);
  hipMemsetAsync(acc0, 0, (size_t)NMAX0 * 3 * 4, stream);

  // occupancy pyramid + compacted row maps
  mark_points<<<(npts + 255) / 256, 256, 0, stream>>>(coords, npts, map0);
  compact_map<<<(D0 * D0 * D0) / 256, 256, 0, stream>>>(map0, D0 * D0 * D0,
                                                        cnt + 0, idx0);
  scatter_feats<<<(npts + 255) / 256, 256, 0, stream>>>(coords, feats, npts,
                                                        map0, acc0);
  cvt_counted<<<(NMAX0 * 3 + 255) / 256, 256, 0, stream>>>(acc0, f0in, cnt, 0, 3);
  mark_parents<D0><<<(NMAX0 + 255) / 256, 256, 0, stream>>>(idx0, cnt, 0, map1);
  compact_map<<<(D1 * D1 * D1) / 256, 256, 0, stream>>>(map1, D1 * D1 * D1,
                                                        cnt + 1, idx1);
  mark_parents<D1><<<(NMAX0 + 255) / 256, 256, 0, stream>>>(idx1, cnt, 1, map2);
  compact_map<<<(D2 * D2 * D2) / 256, 256, 0, stream>>>(map2, D2 * D2 * D2,
                                                        cnt + 2, idx2);

  // pack weights f32 -> fragment-major f16 (K padded to 32)
  const size_t wpo[6] = {WP1, WP2, WP3, WP4, WP5, WP6};
  const int wkt[6] = {81, 1728, 1728, 1728, 864, 864};
  const int wco[6] = {64, 64, 64, 32, 32, 32};
  for (int i = 0; i < 6; ++i) {
    int total = ((wkt[i] + 31) & ~31) * wco[i];
    pack_w_frag<<<(total + 255) / 256, 256, 0, stream>>>(w[i], wpk + wpo[i],
                                                         wkt[i], wco[i]);
  }

  // conv chain (grid sized at max rows; blocks early-exit on device count)
  dim3 g0(NMAX0 / 16), g2(NMAX2 / 16), g3((D3 * D3 * D3) / 16);

  // conv1: 3->64, stride1 @128^3, out at occupied L0 voxels
  conv_wmma<3, 64, 1, D0, false, false, false><<<g0, 32, 0, stream>>>(
      map0, f0in, idx0, cnt, 0, wpk + WP1, b[0], f0a, nullptr, nullptr);
  // conv2: 64->64, stride2 ->64^3, relu, out at occupied L1 voxels
  conv_wmma<64, 64, 2, D0, true, false, false><<<g0, 32, 0, stream>>>(
      map0, f0a, idx1, cnt, 1, wpk + WP2, b[1], f1a, nullptr, nullptr);
  // conv3: 64->64, stride1 @64^3
  conv_wmma<64, 64, 1, D1, false, false, false><<<g0, 32, 0, stream>>>(
      map1, f1a, idx1, cnt, 1, wpk + WP3, b[2], f1b, nullptr, nullptr);
  // conv4: 64->32, stride2 ->32^3, relu
  conv_wmma<64, 32, 2, D1, true, false, false><<<g2, 32, 0, stream>>>(
      map1, f1b, idx2, cnt, 2, wpk + WP4, b[3], f2a, nullptr, nullptr);
  // conv5: 32->32, stride1 @32^3
  conv_wmma<32, 32, 1, D2, false, false, false><<<g2, 32, 0, stream>>>(
      map2, f2a, idx2, cnt, 2, wpk + WP5, b[4], f2b, nullptr, nullptr);
  // conv6: 32->32, stride2 -> dense 16^3 f32 output with occ3 mask
  conv_wmma<32, 32, 2, D2, false, true, true><<<g3, 32, 0, stream>>>(
      map2, f2b, nullptr, cnt, 2, wpk + WP6, b[5], nullptr, (float*)d_out,
      map2);
}
